// LSTM_net_15788299780847
// MI455X (gfx1250) — matrix operand order repaired and checked by reading, not compiled, not run on hardware
//
#include <hip/hip_runtime.h>

// ---------------------------------------------------------------------------
// 2-layer LSTM (B=256, T=512, I=64, H=256) + MLP head for MI455X (gfx1250).
// Batch-tile persistent recurrence: one WG per 16 batch rows, wave w owns
// H-cols [16w,16w+16) and the matching 4 gate tiles -> wave-private cell math.
// bf16 WMMA 16x16x32 with fp32 accumulate; h double-buffered in LDS (1 barrier
// per step); x_{t+1} prefetched during step t (async-to-LDS when available);
// hardware v_tanh for all gate nonlinearities; weights (~2MB bf16) L2-resident.
// ---------------------------------------------------------------------------

typedef __attribute__((ext_vector_type(16))) __bf16 v16bf;
typedef __attribute__((ext_vector_type(8)))  float  v8f;
typedef __attribute__((ext_vector_type(4)))  int    v4i;

namespace {
constexpr int kT   = 512;   // sequence length
constexpr int kH   = 256;   // hidden size
constexpr int kG   = 1024;  // 4*H gate columns
constexpr int kL1  = 128;   // head hidden
constexpr int kOut = 96;    // OUT*NF = 24*4
}

// ---- gfx1250 feature probes (safe fallbacks if builtin absent) -------------
#if defined(__AMDGCN__) && __has_builtin(__builtin_amdgcn_global_load_async_to_lds_b128)
#define HAS_ASYNC_LDS 1
#else
#define HAS_ASYNC_LDS 0
#endif

__device__ __forceinline__ __bf16 f2bf(float f) {
  unsigned u = __builtin_bit_cast(unsigned, f);
  unsigned r = u + 0x7FFFu + ((u >> 16) & 1u);   // round-to-nearest-even
  unsigned short h = (unsigned short)(r >> 16);
  return __builtin_bit_cast(__bf16, h);
}

// Hardware tanh (CDNA5 TRANS32 op) with branch-free fallback.
__device__ __forceinline__ float fast_tanh(float x) {
#if defined(__AMDGCN__) && __has_builtin(__builtin_amdgcn_tanhf)
  return __builtin_amdgcn_tanhf(x);
#elif defined(__AMDGCN__) && __has_builtin(__builtin_amdgcn_tanh_f32)
  return __builtin_amdgcn_tanh_f32(x);
#else
  float ax = fabsf(x);
  float e  = __expf(-2.0f * ax);
  float r  = (1.0f - e) / (1.0f + e);
  return copysignf(r, x);
#endif
}

__device__ __forceinline__ float sigm(float x) {
  return 0.5f + 0.5f * fast_tanh(0.5f * x);      // single trans op on gfx1250
}

// A fragment: 16 x 32 tile of a row-major 16 x ld bf16 matrix at k-base kb.
__device__ __forceinline__ v16bf load_A(const __bf16* base, int ld, int kb, int lane) {
  const int m = lane & 15, half = lane >> 4;
  const __bf16* p0 = base + m * ld + kb + half * 8;
  v16bf out;
#pragma unroll
  for (int i = 0; i < 8; ++i) { out[i] = p0[i]; out[8 + i] = p0[16 + i]; }
  return out;
}

// B fragment: 32 x 16 tile. W stored row-major (N_total x K): row = gate col.
__device__ __forceinline__ v16bf load_B(const __bf16* W, int ld, int ncol, int kb, int lane) {
  const int n = lane & 15, half = lane >> 4;
  const __bf16* p = W + (size_t)(ncol + n) * ld + kb + half * 8;
  v16bf out;
#pragma unroll
  for (int i = 0; i < 8; ++i) { out[i] = p[i]; out[8 + i] = p[16 + i]; }
  return out;
}

__device__ __forceinline__ v8f wmma_bf16(v16bf a, v16bf b, v8f c) {
  return __builtin_amdgcn_wmma_f32_16x16x32_bf16(false, a, false, b, (short)0, c,
                                                 false, false);
}

#if HAS_ASYNC_LDS
// Param 0 is v4i* in the device (global, AS1) address space per clang's
// diagnostic; param 1 is the LDS (AS3) destination.
__device__ __forceinline__ void async_copy16(const void* g, void* l) {
  __builtin_amdgcn_global_load_async_to_lds_b128(
      (__attribute__((address_space(1))) v4i*)(void*)(g ? (void*)g : (void*)g),
      (__attribute__((address_space(3))) v4i*)l, 0, 0);
}
#endif

// ---------------------------------------------------------------------------
// Weight prep
// ---------------------------------------------------------------------------
__global__ void cvt_f32_bf16(const float* __restrict__ src, __bf16* __restrict__ dst,
                             int n) {
  int i = blockIdx.x * blockDim.x + threadIdx.x;
  if (i < n) dst[i] = f2bf(src[i]);
}

__global__ void combine_bias(const float* __restrict__ bih, const float* __restrict__ bhh,
                             float* __restrict__ dst, int n) {
  int i = blockIdx.x * blockDim.x + threadIdx.x;
  if (i < n) dst[i] = bih[i] + bhh[i];
}

// ---------------------------------------------------------------------------
// Fused LSTM layer. Grid = B/16 workgroups, 512 threads (16 waves).
// ---------------------------------------------------------------------------
template <int KIN, bool IN_F32, bool STORE_SEQ, bool STORE_LAST>
__global__ __launch_bounds__(512) void lstm_layer_kernel(
    const void* __restrict__ in_, const __bf16* __restrict__ Wih,
    const __bf16* __restrict__ Whh, const float* __restrict__ bias,
    __bf16* __restrict__ hseq, float* __restrict__ lastout) {
  __shared__ __bf16 hbuf[2][16 * kH];   // double-buffered h_t, 16 KB
  __shared__ __bf16 xbuf[2][16 * kH];   // double-buffered input tile, 16 KB

  const int tid  = threadIdx.x;
  const int lane = tid & 31;
  const int w    = tid >> 5;            // wave 0..15: owns H cols [16w,16w+16)
  const int bt   = blockIdx.x;          // batch tile
  const int n    = lane & 15;
  const int half = lane >> 4;

  // Warm L2 with the (shared, small) weight matrices.
  {
    const char* p = (const char*)Whh;
    for (size_t o = (size_t)tid * 128; o < (size_t)kG * kH * 2; o += 512u * 128u)
      __builtin_prefetch(p + o, 0, 1);
    const char* q = (const char*)Wih;
    for (size_t o = (size_t)tid * 128; o < (size_t)kG * KIN * 2; o += 512u * 128u)
      __builtin_prefetch(q + o, 0, 1);
  }

  const __bf16 bz = __builtin_bit_cast(__bf16, (unsigned short)0);
  for (int i = tid; i < 16 * kH; i += 512) hbuf[0][i] = bz;

  // Stage x tile for t = 0 into xbuf[0].
  if (IN_F32) {
    const float* x = (const float*)in_;
    for (int j = tid; j < 16 * KIN; j += 512) {
      int r = j / KIN, d = j % KIN;
      xbuf[0][r * kH + d] = f2bf(x[(size_t)(bt * 16 + r) * kT * KIN + d]);
    }
  } else {
    const __bf16* x = (const __bf16*)in_;
    for (int j = tid; j < 16 * KIN; j += 512) {
      int r = j / KIN, d = j % KIN;
      xbuf[0][r * kH + d] = x[(size_t)(bt * 16 + r) * kT * KIN + d];
    }
  }
  __syncthreads();

  // Gate biases for this wave's columns (C/D layout broadcasts along M).
  const float b_i = bias[0 * kH + w * 16 + n];
  const float b_f = bias[1 * kH + w * 16 + n];
  const float b_g = bias[2 * kH + w * 16 + n];
  const float b_o = bias[3 * kH + w * 16 + n];

  // Layer 0: hoist the full W_ih fragment set into registers (8 frags, 64 VGPRs).
  v16bf wih0[8];
  if constexpr (KIN == 64) {
#pragma unroll
    for (int kt = 0; kt < 2; ++kt)
#pragma unroll
      for (int q = 0; q < 4; ++q)
        wih0[kt * 4 + q] = load_B(Wih, KIN, q * kH + w * 16, kt * 32, lane);
  }

  v8f c;                                 // cell state for this wave's 16x16 tile
#pragma unroll
  for (int r = 0; r < 8; ++r) c[r] = 0.0f;

  float  xpf[2];                         // fp32 prefetch regs (layer 0)
  __bf16 xpb[8];                         // bf16 prefetch regs (layer 1 fallback)

  for (int t = 0; t < kT; ++t) {
    const int cur = t & 1, nxt = cur ^ 1;
    const bool pf = (t + 1 < kT);        // block-uniform

    // ---- 1. start prefetch of x_{t+1} (overlaps with WMMA below) ----
    if constexpr (IN_F32) {
      if (pf) {
        const float* x = (const float*)in_;
#pragma unroll
        for (int s = 0; s < 2; ++s) {
          int j = tid + s * 512, r = j >> 6, d = j & 63;
          xpf[s] = x[((size_t)(bt * 16 + r) * kT + (t + 1)) * KIN + d];
        }
      }
    } else {
      const __bf16* x = (const __bf16*)in_;
      const int r = tid >> 5, d = (tid & 31) * 8;      // 8 contiguous bf16 (16B)
      const __bf16* src = x + ((size_t)(bt * 16 + r) * kT + (t + 1)) * KIN + d;
#if HAS_ASYNC_LDS
      if (pf) async_copy16((const void*)src, (void*)&xbuf[nxt][r * kH + d]);
#else
      if (pf) {
#pragma unroll
        for (int s = 0; s < 8; ++s) xpb[s] = src[s];
      }
#endif
    }

    // ---- 2. gates = bias + x_t @ Wih^T + h_{t-1} @ Whh^T ----
    v8f ai, af, ag, ao;
#pragma unroll
    for (int r = 0; r < 8; ++r) { ai[r] = b_i; af[r] = b_f; ag[r] = b_g; ao[r] = b_o; }

#pragma unroll
    for (int kt = 0; kt < KIN / 32; ++kt) {
      v16bf a = load_A(xbuf[cur], kH, kt * 32, lane);
      if constexpr (KIN == 64) {
        ai = wmma_bf16(a, wih0[kt * 4 + 0], ai);
        af = wmma_bf16(a, wih0[kt * 4 + 1], af);
        ag = wmma_bf16(a, wih0[kt * 4 + 2], ag);
        ao = wmma_bf16(a, wih0[kt * 4 + 3], ao);
      } else {
        ai = wmma_bf16(a, load_B(Wih, KIN, 0 * kH + w * 16, kt * 32, lane), ai);
        af = wmma_bf16(a, load_B(Wih, KIN, 1 * kH + w * 16, kt * 32, lane), af);
        ag = wmma_bf16(a, load_B(Wih, KIN, 2 * kH + w * 16, kt * 32, lane), ag);
        ao = wmma_bf16(a, load_B(Wih, KIN, 3 * kH + w * 16, kt * 32, lane), ao);
      }
    }
#pragma unroll
    for (int kt = 0; kt < kH / 32; ++kt) {
      v16bf a = load_A(hbuf[cur], kH, kt * 32, lane);
      ai = wmma_bf16(a, load_B(Whh, kH, 0 * kH + w * 16, kt * 32, lane), ai);
      af = wmma_bf16(a, load_B(Whh, kH, 1 * kH + w * 16, kt * 32, lane), af);
      ag = wmma_bf16(a, load_B(Whh, kH, 2 * kH + w * 16, kt * 32, lane), ag);
      ao = wmma_bf16(a, load_B(Whh, kH, 3 * kH + w * 16, kt * 32, lane), ao);
    }

    // ---- 3. land register-prefetched x_{t+1} into xbuf[nxt] ----
    if constexpr (IN_F32) {
      if (pf) {
#pragma unroll
        for (int s = 0; s < 2; ++s) {
          int j = tid + s * 512, r = j >> 6, d = j & 63;
          xbuf[nxt][r * kH + d] = f2bf(xpf[s]);
        }
      }
    } else {
#if !HAS_ASYNC_LDS
      if (pf) {
        const int r = tid >> 5, d = (tid & 31) * 8;
#pragma unroll
        for (int s = 0; s < 8; ++s) xbuf[nxt][r * kH + d + s] = xpb[s];
      }
#endif
    }

    // ---- 4. wave-private cell update; write h_t into hbuf[nxt] ----
#pragma unroll
    for (int r = 0; r < 8; ++r) {
      float iv = sigm(ai[r]);
      float fv = sigm(af[r]);
      float gv = fast_tanh(ag[r]);
      float ov = sigm(ao[r]);
      float cn = fv * c[r] + iv * gv;
      c[r] = cn;
      float hn = ov * fast_tanh(cn);
      const int m   = r + 8 * half;       // C/D layout row
      const int col = w * 16 + n;         // C/D layout col
      const __bf16 hb = f2bf(hn);
      hbuf[nxt][m * kH + col] = hb;
      if (STORE_SEQ)
        hseq[((size_t)(bt * 16 + m) * kT + t) * kH + col] = hb;
      if (STORE_LAST && t == kT - 1)
        lastout[(size_t)(bt * 16 + m) * kH + col] = hn;
    }

#if HAS_ASYNC_LDS
    if constexpr (!IN_F32) asm volatile("s_wait_asynccnt 0" ::: "memory");
#endif
    __syncthreads();   // h_t / x_{t+1} visible to all waves for step t+1
  }
}

// ---------------------------------------------------------------------------
// Head: out = tanh(last @ W1^T + b1) @ W2^T + b2.  Grid = B/16, 256 threads.
// ---------------------------------------------------------------------------
__global__ __launch_bounds__(256) void head_kernel(
    const float* __restrict__ last, const __bf16* __restrict__ W1b,
    const float* __restrict__ b1, const __bf16* __restrict__ W2b,
    const float* __restrict__ b2, float* __restrict__ out) {
  __shared__ __bf16 abuf[16 * kH];
  __shared__ __bf16 zbuf[16 * kL1];

  const int tid  = threadIdx.x;
  const int lane = tid & 31;
  const int w    = tid >> 5;
  const int bt   = blockIdx.x;
  const int n    = lane & 15;
  const int half = lane >> 4;

  for (int j = tid; j < 16 * kH; j += 256)
    abuf[j] = f2bf(last[(size_t)(bt * 16 + (j >> 8)) * kH + (j & 255)]);
  __syncthreads();

  {  // z tile w (16 x 16), K = 256
    v8f acc;
    const float bv = b1[w * 16 + n];
#pragma unroll
    for (int r = 0; r < 8; ++r) acc[r] = bv;
#pragma unroll
    for (int kt = 0; kt < kH / 32; ++kt) {
      v16bf a = load_A(abuf, kH, kt * 32, lane);
      acc = wmma_bf16(a, load_B(W1b, kH, w * 16, kt * 32, lane), acc);
    }
#pragma unroll
    for (int r = 0; r < 8; ++r) {
      const int m = r + 8 * half;
      zbuf[m * kL1 + w * 16 + n] = f2bf(fast_tanh(acc[r]));
    }
  }
  __syncthreads();

  if (w < 6) {  // 6 out tiles of 16 = 96 cols; wave-uniform branch
    v8f acc;
    const float bv = b2[w * 16 + n];
#pragma unroll
    for (int r = 0; r < 8; ++r) acc[r] = bv;
#pragma unroll
    for (int kt = 0; kt < kL1 / 32; ++kt) {
      v16bf a = load_A(zbuf, kL1, kt * 32, lane);
      acc = wmma_bf16(a, load_B(W2b, kL1, w * 16, kt * 32, lane), acc);
    }
#pragma unroll
    for (int r = 0; r < 8; ++r) {
      const int m = r + 8 * half;
      out[(size_t)(bt * 16 + m) * kOut + w * 16 + n] = acc[r];
    }
  }
}

// ---------------------------------------------------------------------------
// Host launcher
// ---------------------------------------------------------------------------
extern "C" void kernel_launch(void* const* d_in, const int* in_sizes, int n_in,
                              void* d_out, int out_size, void* d_ws, size_t ws_size,
                              hipStream_t stream) {
  (void)in_sizes; (void)n_in; (void)out_size; (void)ws_size;

  const float* x     = (const float*)d_in[0];
  const float* W_ih0 = (const float*)d_in[1];
  const float* W_hh0 = (const float*)d_in[2];
  const float* b_ih0 = (const float*)d_in[3];
  const float* b_hh0 = (const float*)d_in[4];
  const float* W_ih1 = (const float*)d_in[5];
  const float* W_hh1 = (const float*)d_in[6];
  const float* b_ih1 = (const float*)d_in[7];
  const float* b_hh1 = (const float*)d_in[8];
  const float* W1    = (const float*)d_in[9];
  const float* b1    = (const float*)d_in[10];
  const float* W2    = (const float*)d_in[11];
  const float* b2    = (const float*)d_in[12];
  float* out = (float*)d_out;

  char* ws = (char*)d_ws;
  __bf16* Wih0b = (__bf16*)(ws + 0);          //  1024*64*2  = 131072
  __bf16* Whh0b = (__bf16*)(ws + 131072);     //  1024*256*2 = 524288
  __bf16* Wih1b = (__bf16*)(ws + 655360);     //  524288
  __bf16* Whh1b = (__bf16*)(ws + 1179648);    //  524288
  __bf16* W1b   = (__bf16*)(ws + 1703936);    //  128*256*2  = 65536
  __bf16* W2b   = (__bf16*)(ws + 1769472);    //  96*128*2   = 24576
  float*  bias0 = (float*)(ws + 1794048);     //  1024*4     = 4096
  float*  bias1 = (float*)(ws + 1798144);     //  4096
  __bf16* h0    = (__bf16*)(ws + 1802240);    //  256*512*256*2 = 67108864
  float*  lastb = (float*)(ws + 68911104);    //  256*256*4  = 262144
  // total ~ 66 MB

  auto cvt = [&](const float* s, __bf16* d, int n) {
    cvt_f32_bf16<<<(n + 255) / 256, 256, 0, stream>>>(s, d, n);
  };
  cvt(W_ih0, Wih0b, 1024 * 64);
  cvt(W_hh0, Whh0b, 1024 * 256);
  cvt(W_ih1, Wih1b, 1024 * 256);
  cvt(W_hh1, Whh1b, 1024 * 256);
  cvt(W1,    W1b,   128 * 256);
  cvt(W2,    W2b,   96 * 128);
  combine_bias<<<4, 256, 0, stream>>>(b_ih0, b_hh0, bias0, 1024);
  combine_bias<<<4, 256, 0, stream>>>(b_ih1, b_hh1, bias1, 1024);

  // Layer 0: x (fp32, KIN=64) -> h0 sequence (bf16)
  lstm_layer_kernel<64, true, true, false><<<16, 512, 0, stream>>>(
      (const void*)x, Wih0b, Whh0b, bias0, h0, nullptr);

  // Layer 1: h0 (bf16, KIN=256) -> last hidden only (fp32)
  lstm_layer_kernel<256, false, false, true><<<16, 512, 0, stream>>>(
      (const void*)h0, Wih1b, Whh1b, bias1, nullptr, lastb);

  // Head
  head_kernel<<<16, 256, 0, stream>>>(lastb, W1b, b1, W2b, b2, out);
}
